// Attention_86131274154367
// MI455X (gfx1250) — matrix operand order repaired and checked by reading, not compiled, not run on hardware
//
#include <hip/hip_runtime.h>
#include <hip/hip_bf16.h>

#define HEADS 8
#define DK 32
#define DV 64
#define BATCH 16
#define FMAP 32
#define NTOK 1024      // FMAP*FMAP
#define CIN 256
#define IDK 256        // HEADS*DK
#define IDV 512        // HEADS*DV
#define QKVD 1024      // IDK+IDK+IDV
#define MROWS (BATCH*NTOK)   // 16384

typedef __attribute__((ext_vector_type(16))) __bf16 v16bf;
typedef __attribute__((ext_vector_type(8)))  float  v8f;
typedef __attribute__((ext_vector_type(4)))  unsigned int u32x4;
typedef __attribute__((ext_vector_type(8)))  int i32x8;
typedef __attribute__((ext_vector_type(4)))  int i32x4;

// native f32 -> bf16 convert (backend emits v_cvt; rounding mode irrelevant here)
static __device__ __forceinline__ __bf16 f2bf(float f) { return (__bf16)f; }

static __device__ __forceinline__ v8f wmma_bf16(v16bf a, v16bf b, v8f c) {
  // 8 args: (neg_a, A, neg_b, B, c_mod, C, reuse_a, reuse_b)
  return __builtin_amdgcn_wmma_f32_16x16x32_bf16(false, a, false, b, (short)0, c,
                                                 false, false);
}

// ---- Tensor Data Mover: 2D bf16 tile (tile_d1 rows x tile_d0 elems) -> LDS ----
// D# per ISA 8.3/8.4: group0 = {flags(count=1), lds_addr, global_addr lo,
// global_addr hi | type=2<<30}; group1 = dims/strides; groups 2,3 = 0 (2D).
// clang-23 toolchain: 6-arg builtin (u32x4, i32x8, i32x4, i32x4, i32x8, i32 cpol).
static __device__ __forceinline__ void tdm_load_2d_bf16(
    unsigned lds_addr, const void* gptr,
    unsigned tile_d0, unsigned tile_d1,
    unsigned tens_d0, unsigned tens_d1,
    unsigned stride0_elems) {
  unsigned long long ga = (unsigned long long)gptr;
  u32x4 g0;
  g0[0] = 1u;                                   // count=1 valid descriptor
  g0[1] = lds_addr;                             // LDS byte address
  g0[2] = (unsigned)ga;                         // global_addr[31:0]
  g0[3] = (unsigned)(ga >> 32) | (2u << 30);    // global_addr[56:32] | type=2
  i32x8 g1;
  g1[0] = (int)(1u << 16);                      // data_size=1 (2 bytes/elem)
  g1[1] = (int)((tens_d0 & 0xffffu) << 16);     // tensor_dim0[15:0]
  g1[2] = (int)((tens_d0 >> 16) | (tens_d1 << 16));   // dim0 hi | dim1 lo
  g1[3] = (int)((tens_d1 >> 16) | (tile_d0 << 16));   // dim1 hi | tile_dim0
  g1[4] = (int)tile_d1;                         // tile_dim1 | tile_dim2=0
  g1[5] = (int)stride0_elems;                   // tensor_dim0_stride[31:0]
  g1[6] = 0;                                    // stride hi | dim1_stride lo
  g1[7] = 0;
  i32x4 z4 = {0, 0, 0, 0};
  i32x8 z8 = {0, 0, 0, 0, 0, 0, 0, 0};
  __builtin_amdgcn_tensor_load_to_lds(g0, g1, z4, z4, z8, 0);
}

// ---- fragment loaders (ISA 7.12.2 wave32 layouts) ----
// A 16x32 (MxK): lanes 0-15 row=lane hold K 0..7 (elems 0-7) and 16..23 (8-15);
// lanes 16-31 same row hold K 8..15 and 24..31. element (m,k) at p + m*sm + k*sk.
static __device__ __forceinline__ v16bf load_a_bf16(const __bf16* p, int sm, int sk, int lane) {
  const int row = lane & 15;
  const int k0 = (lane & 16) ? 8 : 0;
  const __bf16* pr = p + (size_t)row * sm;
  v16bf a;
#pragma unroll
  for (int i = 0; i < 8; ++i) a[i] = pr[(k0 + i) * sk];
#pragma unroll
  for (int i = 0; i < 8; ++i) a[8 + i] = pr[(k0 + 16 + i) * sk];
  return a;
}
static __device__ __forceinline__ v16bf load_a_f32(const float* p, int sm, int sk, int lane) {
  const int row = lane & 15;
  const int k0 = (lane & 16) ? 8 : 0;
  const float* pr = p + (size_t)row * sm;
  v16bf a;
#pragma unroll
  for (int i = 0; i < 8; ++i) a[i] = f2bf(pr[(k0 + i) * sk]);
#pragma unroll
  for (int i = 0; i < 8; ++i) a[8 + i] = f2bf(pr[(k0 + 16 + i) * sk]);
  return a;
}
// B 32x16 (KxN): lanes 0-15 col=lane hold K 0..7 and 16..23; lanes 16-31 hold K 8..15 and 24..31.
// element (k,n) at p + k*sk + n*sn.
static __device__ __forceinline__ v16bf load_b_bf16(const __bf16* p, int sk, int sn, int lane) {
  const int col = lane & 15;
  const int k0 = (lane & 16) ? 8 : 0;
  const __bf16* pc = p + (size_t)col * sn;
  v16bf b;
#pragma unroll
  for (int i = 0; i < 8; ++i) b[i] = pc[(k0 + i) * sk];
#pragma unroll
  for (int i = 0; i < 8; ++i) b[8 + i] = pc[(k0 + 16 + i) * sk];
  return b;
}
static __device__ __forceinline__ v16bf load_b_f32(const float* p, int sk, int sn, int lane) {
  const int col = lane & 15;
  const int k0 = (lane & 16) ? 8 : 0;
  const float* pc = p + (size_t)col * sn;
  v16bf b;
#pragma unroll
  for (int i = 0; i < 8; ++i) b[i] = f2bf(pc[(k0 + i) * sk]);
#pragma unroll
  for (int i = 0; i < 8; ++i) b[8 + i] = f2bf(pc[(k0 + 16 + i) * sk]);
  return b;
}

// reductions across the 16 lanes of a half-wave (rows of a C-tile live in one half)
static __device__ __forceinline__ float hmax16(float v) {
#pragma unroll
  for (int m = 1; m <= 8; m <<= 1) v = fmaxf(v, __shfl_xor(v, m, 32));
  return v;
}
static __device__ __forceinline__ float hsum16(float v) {
#pragma unroll
  for (int m = 1; m <= 8; m <<= 1) v += __shfl_xor(v, m, 32);
  return v;
}

// ============ stage 1: qkv = BN(x @ [w_q|w_k|w_v]); q *= dk^-0.5; bf16 ============
__global__ void __launch_bounds__(256) proj_qkv_kernel(
    const float* __restrict__ x,
    const float* __restrict__ w_q, const float* __restrict__ bn_q,
    const float* __restrict__ w_k, const float* __restrict__ bn_k,
    const float* __restrict__ w_v, const float* __restrict__ bn_v,
    __bf16* __restrict__ qkv) {
  const int lane = threadIdx.x & 31;
  const int wave = threadIdx.x >> 5;
  const int row0 = blockIdx.x * 16;
  const int col0 = (blockIdx.y * 8 + wave) * 16;

  const float* w; const float* bn; int ld; int cseg; float post = 1.0f;
  if (col0 < IDK)            { w = w_q; bn = bn_q; ld = IDK; cseg = col0;
                               post = 0.17677669529663687f; }   // 1/sqrt(DK)
  else if (col0 < 2 * IDK)   { w = w_k; bn = bn_k; ld = IDK; cseg = col0 - IDK; }
  else                       { w = w_v; bn = bn_v; ld = IDV; cseg = col0 - 2 * IDK; }

  v8f acc = {};
#pragma unroll
  for (int kk = 0; kk < CIN; kk += 32) {
    v16bf a = load_a_f32(x + (size_t)row0 * CIN + kk, CIN, 1, lane);
    v16bf b = load_b_f32(w + (size_t)kk * ld + cseg, ld, 1, lane);
    acc = wmma_bf16(a, b, acc);
  }

  const int colt = lane & 15;
  const int roff = (lane & 16) ? 8 : 0;
  const int cc = cseg + colt;
  const float g  = bn[cc];
  const float be = bn[ld + cc];
  const float mu = bn[2 * ld + cc];
  const float vr = bn[3 * ld + cc];
  const float k1 = __frsqrt_rn(vr + 1e-5f) * g;
#pragma unroll
  for (int r = 0; r < 8; ++r) {
    float val = ((acc[r] - mu) * k1 + be) * post;
    qkv[(size_t)(row0 + r + roff) * QKVD + col0 + colt] = f2bf(val);
  }
}

// ============ stage 2: flash attention per (b,h); TDM-fed LDS; GELU fused ============
__global__ void __launch_bounds__(256) attn_kernel(
    const __bf16* __restrict__ qkv, const float* __restrict__ pos_table,
    __bf16* __restrict__ actv) {
  __shared__ __bf16 lds_p[8 * 16 * 32];   // per-wave 16x32 P staging (8 KB)
  __shared__ __bf16 lds_k[32 * 32];       // shared K chunk [token][dk]   (2 KB)
  __shared__ __bf16 lds_v[32 * 64];       // shared V chunk [token][dv]   (4 KB)
  const int lane = threadIdx.x & 31;
  const int wave = threadIdx.x >> 5;
  const int b = blockIdx.x / HEADS;
  const int h = blockIdx.x % HEADS;
  const int q0 = blockIdx.y * 128 + wave * 16;
  const int colt = lane & 15;
  const int roff = (lane & 16) ? 8 : 0;
  const float inv_scale = 5.656854249492381f;   // sqrt(DK): bias added as bias/scale

  const __bf16* base = qkv + (size_t)b * NTOK * QKVD;
  const v16bf qa = load_a_bf16(base + (size_t)q0 * QKVD + h * DK, QKVD, 1, lane);

  const unsigned lds_k_addr = (unsigned)(size_t)&lds_k[0];
  const unsigned lds_v_addr = (unsigned)(size_t)&lds_v[0];

  float m[8], l[8];
#pragma unroll
  for (int r = 0; r < 8; ++r) { m[r] = -1e30f; l[r] = 0.0f; }
  v8f acc0 = {}, acc1 = {}, acc2 = {}, acc3 = {};

  int qri[8], qci[8];
#pragma unroll
  for (int r = 0; r < 8; ++r) { int qi = q0 + r + roff; qri[r] = qi >> 5; qci[r] = qi & 31; }

  __bf16* myp = lds_p + wave * (16 * 32);

  for (int j0 = 0; j0 < NTOK; j0 += 32) {
    // --- TDM: one wave DMAs the shared K/V chunks for this block into LDS ---
    __syncthreads();                      // prior chunk's LDS reads complete
    if (wave == 0) {
      tdm_load_2d_bf16(lds_k_addr, base + (size_t)j0 * QKVD + IDK + h * DK,
                       /*tile*/ DK, 32, /*tensor*/ DK, 32, QKVD);
      tdm_load_2d_bf16(lds_v_addr, base + (size_t)j0 * QKVD + 2 * IDK + h * DV,
                       /*tile*/ DV, 32, /*tensor*/ DV, 32, QKVD);
      __builtin_amdgcn_s_wait_tensorcnt(0);
    }
    __syncthreads();

    // S = Qs @ K^T  (two 16x16 key sub-tiles); B[k=d][n=token] = lds_k[token*32+d]
    v16bf kb0 = load_b_bf16(lds_k, 1, 32, lane);
    v16bf kb1 = load_b_bf16(lds_k + 16 * 32, 1, 32, lane);
    v8f z = {};
    v8f s0 = wmma_bf16(qa, kb0, z);
    v8f s1 = wmma_bf16(qa, kb1, z);

    // relative-position bias
    const int kj0 = j0 + colt,      kr0 = kj0 >> 5, kc0 = kj0 & 31;
    const int kj1 = j0 + 16 + colt, kr1 = kj1 >> 5, kc1 = kj1 & 31;
#pragma unroll
    for (int r = 0; r < 8; ++r) {
      int dr0 = qri[r] - kr0; dr0 = dr0 < 0 ? -dr0 : dr0;
      int dc0 = qci[r] - kc0; dc0 = dc0 < 0 ? -dc0 : dc0;
      int dr1 = qri[r] - kr1; dr1 = dr1 < 0 ? -dr1 : dr1;
      int dc1 = qci[r] - kc1; dc1 = dc1 < 0 ? -dc1 : dc1;
      s0[r] += pos_table[(dr0 * FMAP + dc0) * HEADS + h] * inv_scale;
      s1[r] += pos_table[(dr1 * FMAP + dc1) * HEADS + h] * inv_scale;
    }

    // online softmax update
    float af[8];
#pragma unroll
    for (int r = 0; r < 8; ++r) {
      float mx = hmax16(fmaxf(s0[r], s1[r]));
      float mn = fmaxf(m[r], mx);
      float alpha = __expf(m[r] - mn);
      m[r] = mn;
      float p0 = __expf(s0[r] - mn);
      float p1 = __expf(s1[r] - mn);
      l[r] = l[r] * alpha + hsum16(p0 + p1);
      af[r] = alpha;
      s0[r] = p0; s1[r] = p1;
    }
#pragma unroll
    for (int r = 0; r < 8; ++r) {
      acc0[r] *= af[r]; acc1[r] *= af[r]; acc2[r] *= af[r]; acc3[r] *= af[r];
    }

    // C-layout -> A-layout via per-wave LDS (wave-synchronous; LDS in-order per wave)
#pragma unroll
    for (int r = 0; r < 8; ++r) {
      int row = r + roff;
      myp[row * 32 + colt]      = f2bf(s0[r]);
      myp[row * 32 + 16 + colt] = f2bf(s1[r]);
    }
    v16bf pa = load_a_bf16(myp, 32, 1, lane);

    // O += P @ V; B[k=token][n=dv] = lds_v[token*64 + dv]
    acc0 = wmma_bf16(pa, load_b_bf16(lds_v +  0, DV, 1, lane), acc0);
    acc1 = wmma_bf16(pa, load_b_bf16(lds_v + 16, DV, 1, lane), acc1);
    acc2 = wmma_bf16(pa, load_b_bf16(lds_v + 32, DV, 1, lane), acc2);
    acc3 = wmma_bf16(pa, load_b_bf16(lds_v + 48, DV, 1, lane), acc3);
  }

  // epilogue: normalize, exact GELU, store bf16
#pragma unroll
  for (int r = 0; r < 8; ++r) {
    const int token = q0 + r + roff;
    const float inv_l = 1.0f / l[r];
    float v0 = acc0[r] * inv_l, v1 = acc1[r] * inv_l;
    float v2 = acc2[r] * inv_l, v3 = acc3[r] * inv_l;
    v0 = 0.5f * v0 * (1.0f + erff(v0 * 0.7071067811865475f));
    v1 = 0.5f * v1 * (1.0f + erff(v1 * 0.7071067811865475f));
    v2 = 0.5f * v2 * (1.0f + erff(v2 * 0.7071067811865475f));
    v3 = 0.5f * v3 * (1.0f + erff(v3 * 0.7071067811865475f));
    __bf16* op = actv + ((size_t)b * NTOK + token) * IDV + h * DV + colt;
    op[0]  = f2bf(v0);
    op[16] = f2bf(v1);
    op[32] = f2bf(v2);
    op[48] = f2bf(v3);
  }
}

// ============ stage 3: out = BN(gelu_act @ w_out + b_out), fp32 ============
__global__ void __launch_bounds__(256) out_proj_kernel(
    const __bf16* __restrict__ actv, const float* __restrict__ w_out,
    const float* __restrict__ b_out, const float* __restrict__ bn_out,
    float* __restrict__ out) {
  const int lane = threadIdx.x & 31;
  const int wave = threadIdx.x >> 5;
  const int row0 = blockIdx.x * 16;
  const int col0 = (blockIdx.y * 8 + wave) * 16;

  v8f acc = {};
#pragma unroll
  for (int kk = 0; kk < IDV; kk += 32) {
    v16bf a = load_a_bf16(actv + (size_t)row0 * IDV + kk, IDV, 1, lane);
    v16bf b = load_b_f32(w_out + (size_t)kk * CIN + col0, CIN, 1, lane);
    acc = wmma_bf16(a, b, acc);
  }

  const int colt = lane & 15;
  const int roff = (lane & 16) ? 8 : 0;
  const int cc = col0 + colt;
  const float bias = b_out[cc];
  const float g  = bn_out[cc];
  const float be = bn_out[CIN + cc];
  const float mu = bn_out[2 * CIN + cc];
  const float vr = bn_out[3 * CIN + cc];
  const float k1 = __frsqrt_rn(vr + 1e-5f) * g;
#pragma unroll
  for (int r = 0; r < 8; ++r) {
    float t = acc[r] + bias;
    out[(size_t)(row0 + r + roff) * CIN + cc] = (t - mu) * k1 + be;
  }
}

extern "C" void kernel_launch(void* const* d_in, const int* in_sizes, int n_in,
                              void* d_out, int out_size, void* d_ws, size_t ws_size,
                              hipStream_t stream) {
  const float* x      = (const float*)d_in[0];
  const float* w_q    = (const float*)d_in[1];
  const float* bn_q   = (const float*)d_in[2];
  const float* w_k    = (const float*)d_in[3];
  const float* bn_k   = (const float*)d_in[4];
  const float* w_v    = (const float*)d_in[5];
  const float* bn_v   = (const float*)d_in[6];
  const float* w_out  = (const float*)d_in[7];
  const float* b_out  = (const float*)d_in[8];
  const float* bn_out = (const float*)d_in[9];
  const float* pos    = (const float*)d_in[10];

  // workspace: qkv bf16 [16384,1024] (32 MB) + gelu activations bf16 [16384,512] (16 MB)
  __bf16* qkv  = (__bf16*)d_ws;
  __bf16* actv = qkv + (size_t)MROWS * QKVD;

  proj_qkv_kernel<<<dim3(MROWS / 16, QKVD / 128), 256, 0, stream>>>(
      x, w_q, bn_q, w_k, bn_k, w_v, bn_v, qkv);
  attn_kernel<<<dim3(BATCH * HEADS, NTOK / 128), 256, 0, stream>>>(qkv, pos, actv);
  out_proj_kernel<<<dim3(MROWS / 16, CIN / 128), 256, 0, stream>>>(
      actv, w_out, b_out, bn_out, (float*)d_out);
}
